// UniCrystalFormerLayer_16123307229966
// MI455X (gfx1250) — compile-verified, matrix operand order checked by reading, not run on hardware
//
#include <hip/hip_runtime.h>
#include <stdint.h>

#define DEVINL __device__ __forceinline__

typedef __attribute__((ext_vector_type(16))) __bf16 v16bf;
typedef __attribute__((ext_vector_type(8)))  float  v8f;

static constexpr int   Nn = 16000;
static constexpr int   Ee = 192000;
static constexpr int   Gg = 128;
static constexpr float EPSc    = 1e-5f;
static constexpr float RADIUSc = 5.0f;
static constexpr float PIc     = 3.14159265358979323846f;

// ---------------- packed-weight dword offsets in workspace ----------------
static constexpr size_t OFF_CG1 = 0;        // 192x64 -> 6144 dwords
static constexpr size_t OFF_CG2 = 6144;     // 64x64  -> 2048
static constexpr size_t OFF_CA1 = 8192;
static constexpr size_t OFF_CA2 = 14336;
static constexpr size_t OFF_KU1 = 16384;
static constexpr size_t OFF_KU2 = 22528;
static constexpr size_t OFF_MU1 = 24576;
static constexpr size_t OFF_MU2 = 30720;
static constexpr size_t OFF_WE  = 32768;    // 64x128 -> 4096
static constexpr size_t OFF_WQ  = 36864;
static constexpr size_t OFF_WK  = 40960;
static constexpr size_t OFF_WV  = 45056;
static constexpr size_t OFF_F1  = 49152;
static constexpr size_t OFF_F2  = 53248;    // 128x64 -> 4096
static constexpr size_t OFF_ML  = 57344;
static constexpr size_t OFF_SK  = 59392;
static constexpr size_t WPACK_TOTAL = 61440;

// ---------------- float offsets (workspace viewed as float*) ----------------
static constexpr size_t FB = WPACK_TOTAL;
static constexpr size_t OFF_BNE_SUM = FB + 0;
static constexpr size_t OFF_BNE_SQ  = FB + 64;
static constexpr size_t OFF_BNN_SUM = FB + 128;
static constexpr size_t OFF_BNN_SQ  = FB + 192;
static constexpr size_t OFF_GNC_CNT = FB + 256;
static constexpr size_t OFF_GNC_SUM = FB + 384;
static constexpr size_t OFF_GNC_VAR = FB + 8576;
static constexpr size_t OFF_GNM_CNT = FB + 16768;
static constexpr size_t OFF_GNM_SUM = FB + 16896;
static constexpr size_t OFF_GNM_VAR = FB + 25088;
static constexpr size_t OFF_AGG     = FB + 33280;                       // N*64
static constexpr size_t OFF_OUTSUM  = OFF_AGG + (size_t)Nn*64;          // N*64
static constexpr size_t ZERO_BASE   = FB;
static constexpr size_t ZERO_COUNT  = 33280 + 2*(size_t)Nn*64;
static constexpr size_t OFF_GATE    = OFF_OUTSUM + (size_t)Nn*64;       // E*64
static constexpr size_t OFF_MSG     = OFF_GATE + (size_t)Ee*64;         // E*64
static constexpr size_t OFF_Q       = OFF_MSG + (size_t)Ee*64;          // N*128
static constexpr size_t OFF_K       = OFF_Q   + (size_t)Nn*128;
static constexpr size_t OFF_V       = OFF_K   + (size_t)Nn*128;
static constexpr size_t OFF_XR      = OFF_V   + (size_t)Nn*128;         // N*64
static constexpr size_t OFF_XMA     = OFF_XR  + (size_t)Nn*64;          // N*64
static constexpr size_t OFF_FFNH    = OFF_XMA + (size_t)Nn*64;          // N*128
static constexpr size_t OFF_XMB     = OFF_FFNH+ (size_t)Nn*128;         // N*64
static constexpr size_t OFF_GNT     = OFF_XMB + (size_t)Nn*64;          // N*64
static constexpr size_t WS_FLOATS   = OFF_GNT + (size_t)Nn*64;

// ---------------- helpers ----------------
union AFrag { v16bf v; uint32_t u[8]; };
union BfPair { struct { __bf16 lo, hi; } s; uint32_t u; };

DEVINL uint32_t pack2bf(float a, float b) {
  // native f32->bf16 converts (compiler picks v_cvt / packed cvt)
  BfPair r;
  r.s.lo = (__bf16)a;
  r.s.hi = (__bf16)b;
  return r.u;
}

DEVINL v8f zero8() { v8f z = {0.f,0.f,0.f,0.f,0.f,0.f,0.f,0.f}; return z; }

DEVINL v8f wmma_bf16(const AFrag& a, const AFrag& b, v8f c) {
  return __builtin_amdgcn_wmma_f32_16x16x32_bf16(false, a.v, false, b.v,
                                                 (short)0, c, false, false);
}

DEVINL AFrag load_btile(const uint32_t* __restrict__ wp, int tile, int lane) {
  AFrag f;
  const uint4* p = reinterpret_cast<const uint4*>(wp + (size_t)tile*256 + lane*8);
  uint4 q0 = p[0], q1 = p[1];
  f.u[0]=q0.x; f.u[1]=q0.y; f.u[2]=q0.z; f.u[3]=q0.w;
  f.u[4]=q1.x; f.u[5]=q1.y; f.u[6]=q1.z; f.u[7]=q1.w;
  return f;
}

// A-operand (16x32 bf16) lane layout: lane (m = l&15, g = l>>4) needs
// K = {8g..8g+7} (VGPR0-3) and K = {16+8g..16+8g+7} (VGPR4-7) of row m.
DEVINL AFrag a_from_row(const float* row, int g) {
  AFrag f;
  const float4* p0 = reinterpret_cast<const float4*>(row + 8*g);
  const float4* p1 = reinterpret_cast<const float4*>(row + 16 + 8*g);
  float4 a0 = p0[0], a1 = p0[1], c0 = p1[0], c1 = p1[1];
  f.v[0]  = (__bf16)a0.x; f.v[1]  = (__bf16)a0.y;
  f.v[2]  = (__bf16)a0.z; f.v[3]  = (__bf16)a0.w;
  f.v[4]  = (__bf16)a1.x; f.v[5]  = (__bf16)a1.y;
  f.v[6]  = (__bf16)a1.z; f.v[7]  = (__bf16)a1.w;
  f.v[8]  = (__bf16)c0.x; f.v[9]  = (__bf16)c0.y;
  f.v[10] = (__bf16)c0.z; f.v[11] = (__bf16)c0.w;
  f.v[12] = (__bf16)c1.x; f.v[13] = (__bf16)c1.y;
  f.v[14] = (__bf16)c1.z; f.v[15] = (__bf16)c1.w;
  return f;
}

DEVINL void lds_fence() { asm volatile("s_wait_dscnt 0" ::: "memory"); }
DEVINL float silu_f(float x) { return x / (1.f + __expf(-x)); }

// 2-layer MLP on a 16-row x 192-col A tile: (192->64 silu, 64->64).
// TO_GLOBAL=1: final result to global row-major (ld=64); else to LDS 16x68 buf.
template<int TO_GLOBAL>
DEVINL void mlp2_192(const AFrag (&a)[6],
                     const uint32_t* __restrict__ w1, const uint32_t* __restrict__ w2,
                     const float* __restrict__ b1, const float* __restrict__ b2,
                     float* hid, float* outp, int lane, int rowbase, int R) {
  int mcol = lane & 15, g = lane >> 4;
#pragma unroll
  for (int nt = 0; nt < 4; ++nt) {
    v8f acc = zero8();
#pragma unroll
    for (int c = 0; c < 6; ++c) {
      AFrag b = load_btile(w1, c*4 + nt, lane);
      acc = wmma_bf16(a[c], b, acc);
    }
    float bv = b1[nt*16 + mcol];
#pragma unroll
    for (int j = 0; j < 8; ++j) {
      float x = silu_f(acc[j] + bv);
      hid[(g*8 + j)*68 + nt*16 + mcol] = x;
    }
  }
  lds_fence();
  AFrag h2[2];
  h2[0] = a_from_row(hid + mcol*68,      g);
  h2[1] = a_from_row(hid + mcol*68 + 32, g);
#pragma unroll
  for (int nt = 0; nt < 4; ++nt) {
    v8f acc = zero8();
#pragma unroll
    for (int c = 0; c < 2; ++c) {
      AFrag b = load_btile(w2, c*4 + nt, lane);
      acc = wmma_bf16(h2[c], b, acc);
    }
    float bv = b2[nt*16 + mcol];
#pragma unroll
    for (int j = 0; j < 8; ++j) {
      float x = acc[j] + bv;
      if (TO_GLOBAL) {
        int r = rowbase + g*8 + j;
        if (r < R) outp[(size_t)r*64 + nt*16 + mcol] = x;
      } else {
        outp[(g*8 + j)*68 + nt*16 + mcol] = x;
      }
    }
  }
  if (!TO_GLOBAL) lds_fence();
}

// ---------------- weight packing (fp32 -> bf16 B-operand layout) ----------------
__global__ void pack_kernel(const float* __restrict__ W, uint32_t* __restrict__ outp,
                            int Kdim, int Nout) {
  int total = (Kdim * Nout) >> 1;
  int NTt = Nout >> 4;
  for (int i = blockIdx.x*blockDim.x + threadIdx.x; i < total;
       i += gridDim.x*blockDim.x) {
    int j    = i & 7;
    int lane = (i >> 3) & 31;
    int tile = i >> 8;
    int kc = tile / NTt, nt = tile - kc*NTt;
    int n = nt*16 + (lane & 15);
    int g = lane >> 4;
    int k = kc*32 + ((j < 4) ? (8*g + 2*j) : (16 + 8*g + 2*(j-4)));
    outp[i] = pack2bf(W[(size_t)k*Nout + n], W[(size_t)(k+1)*Nout + n]);
  }
}

// ---------------- generic WMMA projection: out[R,NT*16] = in[R,KC*32]@W + b ----------------
template<int KC, int NT, int ACT>
__global__ void __launch_bounds__(128) proj_kernel(
    const float* __restrict__ in, const uint32_t* __restrict__ wp,
    const float* __restrict__ bias, float* __restrict__ out, int R) {
  int lane = threadIdx.x & 31, wid = threadIdx.x >> 5;
  int tile = blockIdx.x*4 + wid;
  if (tile*16 >= R) return;
  __builtin_prefetch((const void*)wp, 0, 1);
  int mcol = lane & 15, g = lane >> 4;
  int row = tile*16 + mcol; if (row > R-1) row = R-1;
  const float* rp = in + (size_t)row*(KC*32);
  AFrag a[KC];
#pragma unroll
  for (int c = 0; c < KC; ++c) a[c] = a_from_row(rp + 32*c, g);
#pragma unroll
  for (int nt = 0; nt < NT; ++nt) {
    v8f acc = zero8();
#pragma unroll
    for (int c = 0; c < KC; ++c) {
      AFrag b = load_btile(wp, c*NT + nt, lane);
      acc = wmma_bf16(a[c], b, acc);
    }
    float bv = bias[nt*16 + mcol];
#pragma unroll
    for (int j = 0; j < 8; ++j) {
      float x = acc[j] + bv;
      if (ACT) x = silu_f(x);
      int r = tile*16 + g*8 + j;
      if (r < R) out[(size_t)r*(NT*16) + nt*16 + mcol] = x;
    }
  }
}

// ---------------- CartNet fused edge kernel ----------------
__global__ void __launch_bounds__(128) cart_edge_kernel(
    const float* __restrict__ xcart, const float* __restrict__ ecart,
    const int* __restrict__ ei, const uint32_t* __restrict__ wp,
    const float* __restrict__ cg_b1, const float* __restrict__ cg_b2,
    const float* __restrict__ ca_b1, const float* __restrict__ ca_b2,
    float* __restrict__ gate_pre, float* __restrict__ msg_pre) {
  __shared__ __align__(16) float lds[4*16*68];
  int lane = threadIdx.x & 31, wid = threadIdx.x >> 5;
  float* hid = lds + wid*(16*68);
  int tile = blockIdx.x*4 + wid;
  if (tile*16 >= Ee) return;
  __builtin_prefetch((const void*)wp, 0, 1);
  int m = lane & 15, g = lane >> 4;
  int e = tile*16 + m; if (e > Ee-1) e = Ee-1;
  int s = ei[e], d = ei[Ee + e];
  const float* pd = xcart + (size_t)d*64;
  const float* ps = xcart + (size_t)s*64;
  const float* pe = ecart + (size_t)e*64;
  AFrag a[6];
  a[0]=a_from_row(pd,g);    a[1]=a_from_row(pd+32,g);
  a[2]=a_from_row(ps,g);    a[3]=a_from_row(ps+32,g);
  a[4]=a_from_row(pe,g);    a[5]=a_from_row(pe+32,g);
  mlp2_192<1>(a, wp+OFF_CG1, wp+OFF_CG2, cg_b1, cg_b2, hid, gate_pre, lane, tile*16, Ee);
  mlp2_192<1>(a, wp+OFF_CA1, wp+OFF_CA2, ca_b1, ca_b2, hid, msg_pre,  lane, tile*16, Ee);
}

// ---------------- Matformer fused edge kernel (rows = h*E + e, E%16==0) ----------------
__global__ void __launch_bounds__(64) mat_edge_kernel(
    const float* __restrict__ emat, const int* __restrict__ ei,
    const float* __restrict__ q, const float* __restrict__ k, const float* __restrict__ v,
    const uint32_t* __restrict__ wp, const float* __restrict__ be,
    const float* __restrict__ ku_b1, const float* __restrict__ ku_b2,
    const float* __restrict__ mu_b1, const float* __restrict__ mu_b2,
    const float* __restrict__ ml_b,
    const float* __restrict__ lnag, const float* __restrict__ lnab,
    const float* __restrict__ lnmg, const float* __restrict__ lnmb,
    float* __restrict__ out_sum) {
  __shared__ __align__(16) float lds[2*4*16*68];
  int lane = threadIdx.x & 31, wid = threadIdx.x >> 5;
  float* ep  = lds + wid*(4*16*68);
  float* hid = ep  + 16*68;
  float* kub = hid + 16*68;
  float* mub = kub + 16*68;
  int tile = blockIdx.x*2 + wid;
  const int R = 2*Ee;
  if (tile*16 >= R) return;
  __builtin_prefetch((const void*)wp, 0, 1);
  int m = lane & 15, g = lane >> 4;
  int h = (tile*16) / Ee;                  // wave-uniform (E%16==0)
  int r = tile*16 + m; if (r > R-1) r = R-1;
  int e = r - h*Ee;
  int s = ei[e], d = ei[Ee + e];

  // ep = e_mat[e] @ we[:, h*64 : h*64+64] + be
  {
    const float* pe = emat + (size_t)e*64;
    AFrag ae[2]; ae[0]=a_from_row(pe,g); ae[1]=a_from_row(pe+32,g);
#pragma unroll
    for (int nt = 0; nt < 4; ++nt) {
      v8f acc = zero8();
#pragma unroll
      for (int c = 0; c < 2; ++c) {
        AFrag b = load_btile(wp + OFF_WE, c*8 + h*4 + nt, lane);
        acc = wmma_bf16(ae[c], b, acc);
      }
      float bv = be[h*64 + nt*16 + m];
#pragma unroll
      for (int j = 0; j < 8; ++j) ep[(g*8+j)*68 + nt*16 + m] = acc[j] + bv;
    }
  }
  lds_fence();

  AFrag a6[6];
  {
    const float* kd = k + (size_t)d*128 + h*64;
    const float* ks = k + (size_t)s*128 + h*64;
    a6[0]=a_from_row(kd,g);             a6[1]=a_from_row(kd+32,g);
    a6[2]=a_from_row(ks,g);             a6[3]=a_from_row(ks+32,g);
    a6[4]=a_from_row(ep + m*68, g);     a6[5]=a_from_row(ep + m*68 + 32, g);
  }
  mlp2_192<0>(a6, wp+OFF_KU1, wp+OFF_KU2, ku_b1, ku_b2, hid, kub, lane, 0, 0);
  {
    const float* vd = v + (size_t)d*128 + h*64;
    const float* vs = v + (size_t)s*128 + h*64;
    a6[0]=a_from_row(vd,g); a6[1]=a_from_row(vd+32,g);
    a6[2]=a_from_row(vs,g); a6[3]=a_from_row(vs+32,g);
  }
  mlp2_192<0>(a6, wp+OFF_MU1, wp+OFF_MU2, mu_b1, mu_b2, hid, mub, lane, 0, 0);

  // row phase: alpha = q[dst]*ku/8, LayerNorm, sigmoid, mg = mu*sig
  {
    const float* qd = q + (size_t)d*128 + h*64;
    float av[32]; float sum = 0.f, sq = 0.f;
#pragma unroll
    for (int c = 0; c < 32; ++c) {
      int col = g*32 + c;
      float x = qd[col] * kub[m*68 + col] * 0.125f;
      av[c] = x; sum += x; sq += x*x;
    }
    sum += __shfl_xor(sum, 16); sq += __shfl_xor(sq, 16);
    float mean = sum*(1.f/64.f);
    float inv  = rsqrtf(sq*(1.f/64.f) - mean*mean + EPSc);
#pragma unroll
    for (int c = 0; c < 32; ++c) {
      int col = g*32 + c;
      float an = lnag[col]*(av[c]-mean)*inv + lnab[col];
      float sg = 1.f/(1.f+__expf(-an));
      hid[m*68 + col] = mub[m*68 + col] * sg;
    }
  }
  lds_fence();

  // mp = mg @ ml_w + ml_b  (into ep buffer, ep no longer needed)
  {
    AFrag m2[2];
    m2[0]=a_from_row(hid + m*68, g); m2[1]=a_from_row(hid + m*68 + 32, g);
#pragma unroll
    for (int nt = 0; nt < 4; ++nt) {
      v8f acc = zero8();
#pragma unroll
      for (int c = 0; c < 2; ++c) {
        AFrag b = load_btile(wp + OFF_ML, c*4 + nt, lane);
        acc = wmma_bf16(m2[c], b, acc);
      }
      float bv = ml_b[nt*16 + m];
#pragma unroll
      for (int j = 0; j < 8; ++j) ep[(g*8+j)*68 + nt*16 + m] = acc[j] + bv;
    }
  }
  lds_fence();

  // final LayerNorm + head-averaged segment sum
  {
    float mv[32]; float sum = 0.f, sq = 0.f;
#pragma unroll
    for (int c = 0; c < 32; ++c) {
      float x = ep[m*68 + g*32 + c];
      mv[c] = x; sum += x; sq += x*x;
    }
    sum += __shfl_xor(sum, 16); sq += __shfl_xor(sq, 16);
    float mean = sum*(1.f/64.f);
    float inv  = rsqrtf(sq*(1.f/64.f) - mean*mean + EPSc);
#pragma unroll
    for (int c = 0; c < 32; ++c) {
      int col = g*32 + c;
      float y = lnmg[col]*(mv[c]-mean)*inv + lnmb[col];
      atomicAdd(&out_sum[(size_t)d*64 + col], y * 0.5f);   // 1/H, H=2
    }
  }
}

// ---------------- element-wise / reduction kernels ----------------
__global__ void zero_kernel(float* __restrict__ p, size_t n) {
  size_t i = (size_t)blockIdx.x*blockDim.x + threadIdx.x;
  if (i < n) p[i] = 0.f;
}

__global__ void bn_stats_kernel(const float* __restrict__ x, int R,
                                float* __restrict__ sum, float* __restrict__ sq) {
  int gid = blockIdx.x*blockDim.x + threadIdx.x;
  int c = gid & 63;
  int slice = gid >> 6;
  int nsl = (gridDim.x*blockDim.x) >> 6;
  float s = 0.f, q2 = 0.f;
  for (int r = slice; r < R; r += nsl) {
    float v0 = x[(size_t)r*64 + c];
    s += v0; q2 += v0*v0;
  }
  atomicAdd(&sum[c], s);
  atomicAdd(&sq[c], q2);
}

__global__ void cart_apply_kernel(const float* __restrict__ gate_pre,
    const float* __restrict__ msg_pre, const float* __restrict__ dist,
    const int* __restrict__ ei, const float* __restrict__ bsum,
    const float* __restrict__ bsq, const float* __restrict__ bg,
    const float* __restrict__ bb, float* __restrict__ agg) {
  int i = blockIdx.x*blockDim.x + threadIdx.x;
  if (i >= Ee*64) return;
  int e = i >> 6, c = i & 63;
  float invR = 1.f/(float)Ee;
  float mean = bsum[c]*invR;
  float var  = bsq[c]*invR - mean*mean;
  float gn = bg[c]*(gate_pre[i]-mean)*rsqrtf(var+EPSc) + bb[c];
  float dd = dist[e];
  float env = (dd < RADIUSc) ? 0.5f*(__cosf(PIc*dd/RADIUSc)+1.f) : 0.f;
  float gate = env / (1.f+__expf(-gn));
  atomicAdd(&agg[(size_t)ei[Ee+e]*64 + c], gate*msg_pre[i]);
}

__global__ void cart_node_kernel(const float* __restrict__ agg,
    const float* __restrict__ xin, const float* __restrict__ bsum,
    const float* __restrict__ bsq, const float* __restrict__ bg,
    const float* __restrict__ bb, float* __restrict__ outp) {
  int i = blockIdx.x*blockDim.x + threadIdx.x;
  if (i >= Nn*64) return;
  int c = i & 63;
  float invR = 1.f/(float)Nn;
  float mean = bsum[c]*invR;
  float var  = bsq[c]*invR - mean*mean;
  float x = bg[c]*(agg[i]-mean)*rsqrtf(var+EPSc) + bb[c];
  x = silu_f(x);
  outp[i] = x + xin[i];
}

__global__ void gn_pass1(const float* __restrict__ x, const int* __restrict__ seg,
                         float* __restrict__ cnt, float* __restrict__ sum) {
  int i = blockIdx.x*blockDim.x + threadIdx.x;
  if (i >= Nn*64) return;
  int node = i >> 6, c = i & 63;
  int gi = seg[node];
  atomicAdd(&sum[gi*64 + c], x[i]);
  if (c == 0) atomicAdd(&cnt[gi], 1.f);
}

__global__ void gn_pass2(const float* __restrict__ x, const int* __restrict__ seg,
                         const float* __restrict__ cnt, const float* __restrict__ sum,
                         const float* __restrict__ ms, float* __restrict__ tmp,
                         float* __restrict__ var) {
  int i = blockIdx.x*blockDim.x + threadIdx.x;
  if (i >= Nn*64) return;
  int node = i >> 6, c = i & 63;
  int gi = seg[node];
  float ct = fmaxf(cnt[gi], 1.f);
  float o = x[i] - (sum[gi*64 + c]/ct)*ms[c];
  tmp[i] = o;
  atomicAdd(&var[gi*64 + c], o*o);
}

__global__ void gn_pass3(const float* __restrict__ tmp, const int* __restrict__ seg,
                         const float* __restrict__ cnt, const float* __restrict__ var,
                         const float* __restrict__ w, const float* __restrict__ b,
                         float* __restrict__ outp) {
  int i = blockIdx.x*blockDim.x + threadIdx.x;
  if (i >= Nn*64) return;
  int node = i >> 6, c = i & 63;
  int gi = seg[node];
  float ct = fmaxf(cnt[gi], 1.f);
  outp[i] = w[c]*tmp[i]*rsqrtf(var[gi*64 + c]/ct + EPSc) + b[c];
}

__global__ void beta_mix_kernel(const float* __restrict__ osum,
                                const float* __restrict__ xr,
                                const float* __restrict__ bw,
                                float* __restrict__ xm) {
  int i = blockIdx.x*blockDim.x + threadIdx.x;
  if (i >= Nn) return;
  float acc = 0.f;
  for (int c = 0; c < 64; ++c) {
    float o = osum[(size_t)i*64 + c], rr = xr[(size_t)i*64 + c];
    acc += o*bw[c] + rr*bw[64 + c] + (o - rr)*bw[128 + c];
  }
  float beta = 1.f/(1.f+__expf(-acc));
  for (int c = 0; c < 64; ++c) {
    float o = osum[(size_t)i*64 + c], rr = xr[(size_t)i*64 + c];
    xm[(size_t)i*64 + c] = beta*rr + (1.f - beta)*o;
  }
}

// ---------------- input indices (setup_inputs dict order, params flattened) ----------------
enum {
  IN_XCART = 0, IN_XMAT, IN_ECART, IN_EMAT, IN_DIST, IN_EIDX, IN_BATCH,
  P_CG_W1, P_CG_B1, P_CG_W2, P_CG_B2, P_CA_W1, P_CA_B1, P_CA_W2, P_CA_B2,
  P_BN_E_G, P_BN_E_B, P_BN_X_G, P_BN_X_B,
  P_WQ, P_BQ, P_WK, P_BK, P_WV, P_BV, P_WE, P_BE,
  P_KU_W1, P_KU_B1, P_KU_W2, P_KU_B2, P_MU_W1, P_MU_B1, P_MU_W2, P_MU_B2,
  P_LN_A_G, P_LN_A_B, P_ML_W, P_ML_B, P_LN_M_G, P_LN_M_B,
  P_SKIP_W, P_SKIP_B, P_BETA_W, P_F_W1, P_F_B1, P_F_W2, P_F_B2,
  P_GNC_W, P_GNC_B, P_GNC_MS, P_GNM_W, P_GNM_B, P_GNM_MS
};

extern "C" void kernel_launch(void* const* d_in, const int* in_sizes, int n_in,
                              void* d_out, int out_size, void* d_ws, size_t ws_size,
                              hipStream_t stream) {
  (void)in_sizes; (void)n_in; (void)out_size;
  if (ws_size < WS_FLOATS * sizeof(float)) return;

  const float* xcart = (const float*)d_in[IN_XCART];
  const float* xmat  = (const float*)d_in[IN_XMAT];
  const float* ecart = (const float*)d_in[IN_ECART];
  const float* emat  = (const float*)d_in[IN_EMAT];
  const float* dist  = (const float*)d_in[IN_DIST];
  const int*   ei    = (const int*)d_in[IN_EIDX];
  const int*   batch = (const int*)d_in[IN_BATCH];
  float*    outp = (float*)d_out;
  float*    wsf  = (float*)d_ws;
  uint32_t* wsu  = (uint32_t*)d_ws;

#define PARAM(idx) ((const float*)d_in[idx])

  // 0) zero stats + accumulators
  zero_kernel<<<dim3((unsigned)((ZERO_COUNT + 255)/256)), dim3(256), 0, stream>>>(
      wsf + ZERO_BASE, ZERO_COUNT);

  // 1) pack all weights to bf16 B-operand layout
#define PACK(idx, K, NOUT, off)                                                  \
  pack_kernel<<<dim3((((K)*(NOUT)/2) + 255)/256), dim3(256), 0, stream>>>(       \
      PARAM(idx), wsu + (off), (K), (NOUT))
  PACK(P_CG_W1, 192,  64, OFF_CG1);  PACK(P_CG_W2,  64,  64, OFF_CG2);
  PACK(P_CA_W1, 192,  64, OFF_CA1);  PACK(P_CA_W2,  64,  64, OFF_CA2);
  PACK(P_KU_W1, 192,  64, OFF_KU1);  PACK(P_KU_W2,  64,  64, OFF_KU2);
  PACK(P_MU_W1, 192,  64, OFF_MU1);  PACK(P_MU_W2,  64,  64, OFF_MU2);
  PACK(P_WE,     64, 128, OFF_WE);
  PACK(P_WQ,     64, 128, OFF_WQ);   PACK(P_WK,     64, 128, OFF_WK);
  PACK(P_WV,     64, 128, OFF_WV);
  PACK(P_F_W1,   64, 128, OFF_F1);   PACK(P_F_W2,  128,  64, OFF_F2);
  PACK(P_ML_W,   64,  64, OFF_ML);   PACK(P_SKIP_W, 64,  64, OFF_SK);
#undef PACK

  // 2) CartNet edge MLPs (WMMA) -> gate_pre, msg_pre
  cart_edge_kernel<<<dim3(Ee/16/4), dim3(128), 0, stream>>>(
      xcart, ecart, ei, wsu, PARAM(P_CG_B1), PARAM(P_CG_B2),
      PARAM(P_CA_B1), PARAM(P_CA_B2), wsf + OFF_GATE, wsf + OFF_MSG);

  // 3) BN stats over edges, gate+env apply, segment-sum into agg
  bn_stats_kernel<<<dim3(256), dim3(256), 0, stream>>>(
      wsf + OFF_GATE, Ee, wsf + OFF_BNE_SUM, wsf + OFF_BNE_SQ);
  cart_apply_kernel<<<dim3((Ee*64 + 255)/256), dim3(256), 0, stream>>>(
      wsf + OFF_GATE, wsf + OFF_MSG, dist, ei, wsf + OFF_BNE_SUM, wsf + OFF_BNE_SQ,
      PARAM(P_BN_E_G), PARAM(P_BN_E_B), wsf + OFF_AGG);

  // 4) BN(agg) + silu + residual -> xc (pre-GN) stored in d_out[0:N*64]
  bn_stats_kernel<<<dim3(256), dim3(256), 0, stream>>>(
      wsf + OFF_AGG, Nn, wsf + OFF_BNN_SUM, wsf + OFF_BNN_SQ);
  cart_node_kernel<<<dim3((Nn*64 + 255)/256), dim3(256), 0, stream>>>(
      wsf + OFF_AGG, xcart, wsf + OFF_BNN_SUM, wsf + OFF_BNN_SQ,
      PARAM(P_BN_X_G), PARAM(P_BN_X_B), outp);

  // 5) GraphNorm on xc (in-place on d_out)
  {
    dim3 gdim((Nn*64 + 255)/256), bdim(256);
    gn_pass1<<<gdim, bdim, 0, stream>>>(outp, batch, wsf + OFF_GNC_CNT, wsf + OFF_GNC_SUM);
    gn_pass2<<<gdim, bdim, 0, stream>>>(outp, batch, wsf + OFF_GNC_CNT, wsf + OFF_GNC_SUM,
                                        PARAM(P_GNC_MS), wsf + OFF_GNT, wsf + OFF_GNC_VAR);
    gn_pass3<<<gdim, bdim, 0, stream>>>(wsf + OFF_GNT, batch, wsf + OFF_GNC_CNT,
                                        wsf + OFF_GNC_VAR, PARAM(P_GNC_W), PARAM(P_GNC_B), outp);
  }

  // 6) Matformer projections q,k,v (WMMA)
  proj_kernel<2,8,0><<<dim3(Nn/64), dim3(128), 0, stream>>>(
      xmat, wsu + OFF_WQ, PARAM(P_BQ), wsf + OFF_Q, Nn);
  proj_kernel<2,8,0><<<dim3(Nn/64), dim3(128), 0, stream>>>(
      xmat, wsu + OFF_WK, PARAM(P_BK), wsf + OFF_K, Nn);
  proj_kernel<2,8,0><<<dim3(Nn/64), dim3(128), 0, stream>>>(
      xmat, wsu + OFF_WV, PARAM(P_BV), wsf + OFF_V, Nn);

  // 7) fused Matformer attention edge kernel -> out_sum (head-averaged)
  mat_edge_kernel<<<dim3((2*Ee/16)/2), dim3(64), 0, stream>>>(
      emat, ei, wsf + OFF_Q, wsf + OFF_K, wsf + OFF_V, wsu, PARAM(P_BE),
      PARAM(P_KU_B1), PARAM(P_KU_B2), PARAM(P_MU_B1), PARAM(P_MU_B2), PARAM(P_ML_B),
      PARAM(P_LN_A_G), PARAM(P_LN_A_B), PARAM(P_LN_M_G), PARAM(P_LN_M_B),
      wsf + OFF_OUTSUM);

  // 8) skip projection, beta gate mix
  proj_kernel<2,4,0><<<dim3(Nn/64), dim3(128), 0, stream>>>(
      xmat, wsu + OFF_SK, PARAM(P_SKIP_B), wsf + OFF_XR, Nn);
  beta_mix_kernel<<<dim3((Nn + 255)/256), dim3(256), 0, stream>>>(
      wsf + OFF_OUTSUM, wsf + OFF_XR, PARAM(P_BETA_W), wsf + OFF_XMA);

  // 9) FFN (WMMA, silu mid)
  proj_kernel<2,8,1><<<dim3(Nn/64), dim3(128), 0, stream>>>(
      wsf + OFF_XMA, wsu + OFF_F1, PARAM(P_F_B1), wsf + OFF_FFNH, Nn);
  proj_kernel<4,4,0><<<dim3(Nn/64), dim3(128), 0, stream>>>(
      wsf + OFF_FFNH, wsu + OFF_F2, PARAM(P_F_B2), wsf + OFF_XMB, Nn);

  // 10) GraphNorm on xm -> d_out[N*64 : 2*N*64]
  {
    dim3 gdim((Nn*64 + 255)/256), bdim(256);
    gn_pass1<<<gdim, bdim, 0, stream>>>(wsf + OFF_XMB, batch,
                                        wsf + OFF_GNM_CNT, wsf + OFF_GNM_SUM);
    gn_pass2<<<gdim, bdim, 0, stream>>>(wsf + OFF_XMB, batch, wsf + OFF_GNM_CNT,
                                        wsf + OFF_GNM_SUM, PARAM(P_GNM_MS),
                                        wsf + OFF_GNT, wsf + OFF_GNM_VAR);
    gn_pass3<<<gdim, bdim, 0, stream>>>(wsf + OFF_GNT, batch, wsf + OFF_GNM_CNT,
                                        wsf + OFF_GNM_VAR, PARAM(P_GNM_W), PARAM(P_GNM_B),
                                        outp + (size_t)Nn*64);
  }
#undef PARAM
}
